// cVAE_64690797413124
// MI455X (gfx1250) — matrix operand order repaired and checked by reading, not compiled, not run on hardware
//
#include <hip/hip_runtime.h>
#include <hip/hip_bf16.h>
#include <math.h>

// ---------------- types ----------------
typedef __bf16 bf16;
typedef bf16  v8bf  __attribute__((ext_vector_type(8)));
typedef bf16  v16bf __attribute__((ext_vector_type(16)));
typedef float v8f   __attribute__((ext_vector_type(8)));

// ---------------- problem constants ----------------
static constexpr int LSEQ   = 2048;
static constexpr int VOC    = 32000;
static constexpr int HID    = 768;
static constexpr int HCD    = 1024;   // H + C
static constexpr int G4     = 4096;   // 4*HC
static constexpr int TENC   = LSEQ - 2;   // 2046
static constexpr int TDEC   = LSEQ - 1;   // 2047
static constexpr int NCHUNK = 20;     // logits column chunks (1600 cols each)
static constexpr int GPC    = 25;     // 64-col groups per chunk (20*25*64 = 32000)
static constexpr int NWGREC = 32;     // workgroups in recurrence kernels

// ---------------- helpers ----------------
__device__ __forceinline__ float sigf(float x) { return 1.f / (1.f + __expf(-x)); }

__device__ __forceinline__ v8f wmma_bf16(v16bf a, v16bf b, v8f c) {
  return __builtin_amdgcn_wmma_f32_16x16x32_bf16(false, a, false, b, (short)0, c, false, false);
}

// A fragment loader: pa already points at A + row*K + hf*8 (row clamped).
// lane<16 : K in {k0..k0+7} U {k0+16..k0+23}; lane>=16: +8 on both ranges.
__device__ __forceinline__ v16bf ld_a(const bf16* __restrict__ pa, int k0) {
  v8bf lo = *(const v8bf*)(pa + k0);
  v8bf hi = *(const v8bf*)(pa + k0 + 16);
  return __builtin_shufflevector(lo, hi, 0, 1, 2, 3, 4, 5, 6, 7, 8, 9, 10, 11, 12, 13, 14, 15);
}

// B fragment loader: pb already points at W + col*K + hf*8.
// lane<16 : K in {k0..k0+7} U {k0+16..k0+23}; lane>=16: +8 on both ranges.
__device__ __forceinline__ v16bf ld_b(const bf16* __restrict__ pb, int k0) {
  v8bf lo = *(const v8bf*)(pb + k0);
  v8bf hi = *(const v8bf*)(pb + k0 + 16);
  return __builtin_shufflevector(lo, hi, 0, 1, 2, 3, 4, 5, 6, 7, 8, 9, 10, 11, 12, 13, 14, 15);
}

// device-wide sense barrier: bar[0]=count, bar[1]=generation
__device__ __forceinline__ void gbar(int* bar, int n) {
  __syncthreads();
  if (threadIdx.x == 0) {
    __threadfence();
    volatile int* gen = bar + 1;
    int g = *gen;
    if (atomicAdd(bar, 1) == n - 1) {
      bar[0] = 0;
      __threadfence();
      atomicAdd((int*)(bar + 1), 1);
    } else {
      while (*gen == g) { __builtin_amdgcn_s_sleep(1); }
    }
    __threadfence();
  }
  __syncthreads();
}

// ---------------- small utility kernels ----------------
__global__ void k_zero_i32(int* p, int n) {
  int i = blockIdx.x * blockDim.x + threadIdx.x;
  if (i < n) p[i] = 0;
}

__global__ void k_f32_to_bf16(const float* __restrict__ in, bf16* __restrict__ out, size_t n) {
  size_t i = (size_t)blockIdx.x * blockDim.x + threadIdx.x;
  size_t st = (size_t)gridDim.x * blockDim.x;
  for (; i < n; i += st) out[i] = (bf16)in[i];
}

// gather embedding rows (tokens[tok_off + row]) -> bf16 matrix [rows x 1024]
__global__ void k_gather_embed(const float* __restrict__ embed, const int* __restrict__ tokens,
                               int tok_off, bf16* __restrict__ X, int rows) {
  size_t n = (size_t)rows * HCD;
  size_t i = (size_t)blockIdx.x * blockDim.x + threadIdx.x;
  size_t st = (size_t)gridDim.x * blockDim.x;
  for (; i < n; i += st) {
    int row = (int)(i >> 10);
    int d = (int)(i & 1023);
    X[i] = (bf16)embed[(size_t)tokens[tok_off + row] * HCD + d];
  }
}

// ---------------- WMMA GEMM: C[MxN] = A[MxK](bf16) @ W[NxK]^T(bf16) + b0 + b1 ----------------
// Each wave computes a 16x64 strip (4 N-tiles, one shared A fragment),
// K-loop software-pipelined one stage deep.
__global__ __launch_bounds__(256) void k_wmma_gemm_bias(const bf16* __restrict__ A,
                                                        const bf16* __restrict__ W,
                                                        const float* __restrict__ b0,
                                                        const float* __restrict__ b1,
                                                        float* __restrict__ C,
                                                        int M, int N, int K) {
  int lane = threadIdx.x & 31;
  int hf = lane >> 4, r = lane & 15;
  int gw = (blockIdx.x * blockDim.x + threadIdx.x) >> 5;
  int nw = (gridDim.x * blockDim.x) >> 5;
  int tilesM = (M + 15) >> 4;
  int tilesN4 = N >> 6;
  int total = tilesM * tilesN4;
  for (int t = gw; t < total; t += nw) {
    int tm = t / tilesN4, tn = t - tm * tilesN4;
    int m0 = tm << 4, n0 = tn << 6;
    int row = m0 + r; if (row > M - 1) row = M - 1;   // clamp: padded rows never stored
    const bf16* pa  = A + (size_t)row * K + hf * 8;
    const bf16* pb0 = W + (size_t)(n0 + r) * K + hf * 8;
    const bf16* pb1 = pb0 + (size_t)16 * K;
    const bf16* pb2 = pb0 + (size_t)32 * K;
    const bf16* pb3 = pb0 + (size_t)48 * K;
    v8f acc0 = {}, acc1 = {}, acc2 = {}, acc3 = {};
    v16bf a  = ld_a(pa, 0);
    v16bf f0 = ld_b(pb0, 0), f1 = ld_b(pb1, 0), f2 = ld_b(pb2, 0), f3 = ld_b(pb3, 0);
    for (int k0 = 32; k0 < K; k0 += 32) {
      v16bf an = ld_a(pa, k0);
      v16bf g0 = ld_b(pb0, k0), g1 = ld_b(pb1, k0), g2 = ld_b(pb2, k0), g3 = ld_b(pb3, k0);
      acc0 = wmma_bf16(a, f0, acc0);
      acc1 = wmma_bf16(a, f1, acc1);
      acc2 = wmma_bf16(a, f2, acc2);
      acc3 = wmma_bf16(a, f3, acc3);
      a = an; f0 = g0; f1 = g1; f2 = g2; f3 = g3;
    }
    acc0 = wmma_bf16(a, f0, acc0);
    acc1 = wmma_bf16(a, f1, acc1);
    acc2 = wmma_bf16(a, f2, acc2);
    acc3 = wmma_bf16(a, f3, acc3);
    v8f accs[4] = {acc0, acc1, acc2, acc3};
#pragma unroll
    for (int j = 0; j < 4; ++j) {
      int col = n0 + j * 16 + r;
      float bias = 0.f;
      if (b0) bias += b0[col];
      if (b1) bias += b1[col];
#pragma unroll
      for (int i = 0; i < 8; ++i) {
        int orow = m0 + i + hf * 8;
        if (orow < M) C[(size_t)orow * N + col] = accs[j][i] + bias;
      }
    }
  }
}

// ---------------- WMMA logits GEMM fused with chunked log-sum-exp stats ----------------
// Each wave owns a (16-row tile, 1600-col chunk); inner groups of 64 cols share
// one max/sum cross-lane reduction pass.
__global__ __launch_bounds__(256) void k_wmma_logits_lse(const bf16* __restrict__ Hb,
                                                         const bf16* __restrict__ Wo,
                                                         const float* __restrict__ out_b,
                                                         float* __restrict__ chM,
                                                         float* __restrict__ chS,
                                                         int M, int N, int K) {
  int lane = threadIdx.x & 31;
  int hf = lane >> 4, r = lane & 15;
  int gw = (blockIdx.x * blockDim.x + threadIdx.x) >> 5;
  int nw = (gridDim.x * blockDim.x) >> 5;
  int tilesM = (M + 15) >> 4;
  int total = tilesM * NCHUNK;
  for (int task = gw; task < total; task += nw) {
    int tm = task / NCHUNK, ch = task - tm * NCHUNK;
    int m0 = tm << 4;
    int row = m0 + r; if (row > M - 1) row = M - 1;
    const bf16* pa = Hb + (size_t)row * K + hf * 8;
    float rm[8], rs[8];
#pragma unroll
    for (int i = 0; i < 8; ++i) { rm[i] = -3.4e38f; rs[i] = 0.f; }
    for (int g = 0; g < GPC; ++g) {
      int n0 = ch * (GPC * 64) + g * 64;
      const bf16* pb0 = Wo + (size_t)(n0 + r) * K + hf * 8;
      const bf16* pb1 = pb0 + (size_t)16 * K;
      const bf16* pb2 = pb0 + (size_t)32 * K;
      const bf16* pb3 = pb0 + (size_t)48 * K;
      v8f acc0 = {}, acc1 = {}, acc2 = {}, acc3 = {};
      v16bf a  = ld_a(pa, 0);
      v16bf f0 = ld_b(pb0, 0), f1 = ld_b(pb1, 0), f2 = ld_b(pb2, 0), f3 = ld_b(pb3, 0);
      for (int k0 = 32; k0 < K; k0 += 32) {
        v16bf an = ld_a(pa, k0);
        v16bf g0 = ld_b(pb0, k0), g1 = ld_b(pb1, k0), g2 = ld_b(pb2, k0), g3 = ld_b(pb3, k0);
        acc0 = wmma_bf16(a, f0, acc0);
        acc1 = wmma_bf16(a, f1, acc1);
        acc2 = wmma_bf16(a, f2, acc2);
        acc3 = wmma_bf16(a, f3, acc3);
        a = an; f0 = g0; f1 = g1; f2 = g2; f3 = g3;
      }
      acc0 = wmma_bf16(a, f0, acc0);
      acc1 = wmma_bf16(a, f1, acc1);
      acc2 = wmma_bf16(a, f2, acc2);
      acc3 = wmma_bf16(a, f3, acc3);
      float bias0 = out_b[n0 + r];
      float bias1 = out_b[n0 + 16 + r];
      float bias2 = out_b[n0 + 32 + r];
      float bias3 = out_b[n0 + 48 + r];
#pragma unroll
      for (int i = 0; i < 8; ++i) {
        float v0 = acc0[i] + bias0;
        float v1 = acc1[i] + bias1;
        float v2 = acc2[i] + bias2;
        float v3 = acc3[i] + bias3;
        float tmax = fmaxf(fmaxf(v0, v1), fmaxf(v2, v3));
#pragma unroll
        for (int m = 1; m < 16; m <<= 1) tmax = fmaxf(tmax, __shfl_xor(tmax, m, 32));
        float tsum = __expf(v0 - tmax) + __expf(v1 - tmax) + __expf(v2 - tmax) + __expf(v3 - tmax);
#pragma unroll
        for (int m = 1; m < 16; m <<= 1) tsum += __shfl_xor(tsum, m, 32);
        if (tmax > rm[i]) { rs[i] = rs[i] * __expf(rm[i] - tmax) + tsum; rm[i] = tmax; }
        else             { rs[i] += tsum * __expf(tmax - rm[i]); }
      }
    }
    if (r == 0) {
#pragma unroll
      for (int i = 0; i < 8; ++i) {
        int orow = m0 + i + hf * 8;
        if (orow < M) {
          chM[orow * NCHUNK + ch] = rm[i];
          chS[orow * NCHUNK + ch] = rs[i];
        }
      }
    }
  }
}

// ---------------- encoder LSTM recurrence (persistent, 32 WGs, grid barrier) ----------------
__global__ __launch_bounds__(256) void k_lstm_encoder(const float* __restrict__ Whh,
                                                      const float* __restrict__ pre, // T x 4096 incl biases
                                                      const float* __restrict__ cond_embed,
                                                      const int* __restrict__ condition,
                                                      float* __restrict__ h_g, // 2 x 1024 ping-pong
                                                      float* __restrict__ c_g, // 1024 (final cT)
                                                      int T, int* __restrict__ bar) {
  __shared__ float h_s[HCD];
  __shared__ float g_s[128];
  __shared__ float c_s[32];
  int tid = threadIdx.x, lane = tid & 31, wv = tid >> 5;
  int u0 = blockIdx.x * 32;
  const float* ce = cond_embed + condition[0] * 256;
  if (tid < 32) {
    int gu = u0 + tid;
    float v = (gu < HID) ? 0.f : ce[gu - HID];
    c_s[tid] = v;
    h_g[gu] = v;              // h0 into buffer 0
  }
  gbar(bar, NWGREC);
  for (int t = 0; t < T; ++t) {
    const float* hin = h_g + (t & 1) * HCD;
    float* hout = h_g + ((t + 1) & 1) * HCD;
    for (int i = tid; i < HCD; i += 256) h_s[i] = hin[i];
    __syncthreads();
    for (int rr = wv; rr < 128; rr += 8) {
      int q = rr >> 5, j = rr & 31;
      const float* wr = Whh + (size_t)(q * HCD + u0 + j) * HCD;
      float s = 0.f;
      for (int k = lane; k < HCD; k += 32) s = fmaf(wr[k], h_s[k], s);
#pragma unroll
      for (int o = 16; o; o >>= 1) s += __shfl_down(s, o, 32);
      if (lane == 0) g_s[rr] = s + pre[(size_t)t * G4 + q * HCD + u0 + j];
    }
    __syncthreads();
    if (tid < 32) {
      float gi = g_s[tid], gf = g_s[32 + tid], gg = g_s[64 + tid], go = g_s[96 + tid];
      float cn = sigf(gf) * c_s[tid] + sigf(gi) * tanhf(gg);
      float hn = sigf(go) * tanhf(cn);
      c_s[tid] = cn;
      hout[u0 + tid] = hn;
    }
    gbar(bar, NWGREC);
  }
  if (tid < 32) c_g[u0 + tid] = c_s[tid];
}

// ---------------- decoder LSTM recurrence (persistent, 32 WGs, 2 barriers/step) ----------------
__global__ __launch_bounds__(256) void k_lstm_decoder(const float* __restrict__ Whh,
                                                      const float* __restrict__ pre, // T x 4096 incl biases
                                                      const float* __restrict__ fch_w,
                                                      const float* __restrict__ fch_b,
                                                      const float* __restrict__ fcc_w,
                                                      const float* __restrict__ fcc_b,
                                                      const float* __restrict__ cond_embed,
                                                      const int* __restrict__ condition,
                                                      const float* __restrict__ z_h,
                                                      const float* __restrict__ z_c,
                                                      float* __restrict__ h2_g, // 768
                                                      float* __restrict__ c2_g, // 768
                                                      float* __restrict__ hn_g, // 1024
                                                      float* __restrict__ cn_g, // 1024
                                                      bf16* __restrict__ Hall,  // T x 1024
                                                      int T, int* __restrict__ bar) {
  __shared__ float x_s[HCD];
  __shared__ float hn_s[HCD];
  __shared__ float cn_s[HCD];
  __shared__ float g_s[128];
  int tid = threadIdx.x, lane = tid & 31, wv = tid >> 5;
  int wg = blockIdx.x;
  int u0 = wg * 32;
  const float* cd = cond_embed + condition[0] * 256;
  {
    int gi = blockIdx.x * blockDim.x + tid;
    if (gi < HID) { h2_g[gi] = z_h[gi]; c2_g[gi] = z_c[gi]; }
  }
  gbar(bar, NWGREC);
  for (int t = 0; t < T; ++t) {
    // phase A: gates + cell update (produces h_new/c_new, 1024 each)
    for (int i = tid; i < HCD; i += 256) x_s[i] = (i < HID) ? h2_g[i] : cd[i - HID];
    __syncthreads();
    for (int rr = wv; rr < 128; rr += 8) {
      int q = rr >> 5, j = rr & 31;
      const float* wr = Whh + (size_t)(q * HCD + u0 + j) * HCD;
      float s = 0.f;
      for (int k = lane; k < HCD; k += 32) s = fmaf(wr[k], x_s[k], s);
#pragma unroll
      for (int o = 16; o; o >>= 1) s += __shfl_down(s, o, 32);
      if (lane == 0) g_s[rr] = s + pre[(size_t)t * G4 + q * HCD + u0 + j];
    }
    __syncthreads();
    if (tid < 32) {
      int gu = u0 + tid;
      float cc = (gu < HID) ? c2_g[gu] : cd[gu - HID];
      float gi = g_s[tid], gf = g_s[32 + tid], gg = g_s[64 + tid], go = g_s[96 + tid];
      float cn = sigf(gf) * cc + sigf(gi) * tanhf(gg);
      float hn = sigf(go) * tanhf(cn);
      hn_g[gu] = hn;
      cn_g[gu] = cn;
      Hall[(size_t)t * HCD + gu] = (bf16)hn;
    }
    gbar(bar, NWGREC);
    // phase B: h2 = fc_h(h_new), c2 = fc_c(c_new)  (1536 rows over 32 WGs)
    for (int i = tid; i < HCD; i += 256) { hn_s[i] = hn_g[i]; cn_s[i] = cn_g[i]; }
    __syncthreads();
    for (int rr = wv; rr < 48; rr += 8) {
      int fr = wg * 48 + rr;
      const float* wr;
      const float* src;
      float b;
      if (fr < HID) { wr = fch_w + (size_t)fr * HCD; src = hn_s; b = fch_b[fr]; }
      else          { int u = fr - HID; wr = fcc_w + (size_t)u * HCD; src = cn_s; b = fcc_b[u]; }
      float s = 0.f;
      for (int k = lane; k < HCD; k += 32) s = fmaf(wr[k], src[k], s);
#pragma unroll
      for (int o = 16; o; o >>= 1) s += __shfl_down(s, o, 32);
      if (lane == 0) {
        if (fr < HID) h2_g[fr] = s + b;
        else          c2_g[fr - HID] = s + b;
      }
    }
    gbar(bar, NWGREC);
  }
}

// ---------------- VAE head: 4 projections (m_h, lv_h, m_c, lv_c), one wave per row ----------------
__global__ __launch_bounds__(256) void k_fc4(const float* __restrict__ hT, const float* __restrict__ cT,
                                             const float* __restrict__ w0, const float* __restrict__ b0,
                                             const float* __restrict__ w1, const float* __restrict__ b1,
                                             const float* __restrict__ w2, const float* __restrict__ b2,
                                             const float* __restrict__ w3, const float* __restrict__ b3,
                                             float* __restrict__ out) {
  int lane = threadIdx.x & 31;
  int gw = (blockIdx.x * blockDim.x + threadIdx.x) >> 5;
  int nw = (gridDim.x * blockDim.x) >> 5;
  for (int row = gw; row < 4 * HID; row += nw) {
    int q = row / HID, r = row - q * HID;
    const float* src = (q < 2) ? hT : cT;
    const float* w = (q == 0 ? w0 : q == 1 ? w1 : q == 2 ? w2 : w3) + (size_t)r * HCD;
    const float* bb = (q == 0 ? b0 : q == 1 ? b1 : q == 2 ? b2 : b3);
    float s = 0.f;
    for (int k = lane; k < HCD; k += 32) s = fmaf(w[k], src[k], s);
#pragma unroll
    for (int o = 16; o; o >>= 1) s += __shfl_down(s, o, 32);
    if (lane == 0) out[row] = s + bb[r];
  }
}

// ---------------- z sampling + KL terms (single block, deterministic) ----------------
__global__ __launch_bounds__(256) void k_z_kl(const float* __restrict__ proj,
                                              const float* __restrict__ eps_h,
                                              const float* __restrict__ eps_c,
                                              float* __restrict__ z_h, float* __restrict__ z_c,
                                              float* __restrict__ kl) {
  __shared__ float sh[256], sc[256];
  int tid = threadIdx.x;
  float ah = 0.f, ac = 0.f;
  for (int i = tid; i < HID; i += 256) {
    float mh = proj[i], lvh = proj[HID + i], mc = proj[2 * HID + i], lvc = proj[3 * HID + i];
    z_h[i] = eps_h[i] * expf(lvh * 0.5f) + mh;
    z_c[i] = eps_c[i] * expf(lvc * 0.5f) + mc;
    ah += mh * mh + expf(lvh) - 1.f - lvh;
    ac += mc * mc + expf(lvc) - 1.f - lvc;
  }
  sh[tid] = ah; sc[tid] = ac;
  __syncthreads();
  for (int o = 128; o; o >>= 1) {
    if (tid < o) { sh[tid] += sh[tid + o]; sc[tid] += sc[tid + o]; }
    __syncthreads();
  }
  if (tid == 0) { kl[0] = 0.5f * sh[0]; kl[1] = 0.5f * sc[0]; }
}

// ---------------- target logit: logits[t][tgt[t]], one wave per row ----------------
__global__ __launch_bounds__(256) void k_tgt_logit(const bf16* __restrict__ Hb,
                                                   const bf16* __restrict__ Wo,
                                                   const float* __restrict__ out_b,
                                                   const int* __restrict__ tokens,
                                                   float* __restrict__ out, int M) {
  int lane = threadIdx.x & 31;
  int gw = (blockIdx.x * blockDim.x + threadIdx.x) >> 5;
  int nw = (gridDim.x * blockDim.x) >> 5;
  for (int t = gw; t < M; t += nw) {
    int tgt = tokens[t + 1];
    const bf16* h = Hb + (size_t)t * HCD;
    const bf16* w = Wo + (size_t)tgt * HCD;
    float s = 0.f;
    for (int k = lane; k < HCD; k += 32) s = fmaf((float)h[k], (float)w[k], s);
#pragma unroll
    for (int o = 16; o; o >>= 1) s += __shfl_down(s, o, 32);
    if (lane == 0) out[t] = s + out_b[tgt];
  }
}

// ---------------- merge lse chunks, cross-entropy per step ----------------
__global__ void k_ce(const float* __restrict__ chM, const float* __restrict__ chS,
                     const float* __restrict__ tgtl, float* __restrict__ ce, int M) {
  int t = blockIdx.x * blockDim.x + threadIdx.x;
  if (t < M) {
    float m = -3.4e38f, s = 0.f;
    for (int c = 0; c < NCHUNK; ++c) {
      float cm = chM[t * NCHUNK + c], cs = chS[t * NCHUNK + c];
      if (cm > m) { s = s * __expf(m - cm) + cs; m = cm; }
      else        { s += cs * __expf(cm - m); }
    }
    ce[t] = (m + logf(s)) - tgtl[t];
  }
}

// ---------------- final reduction + loss assembly ----------------
__global__ __launch_bounds__(256) void k_final(const float* __restrict__ ce, int M,
                                               const float* __restrict__ kl,
                                               const float* __restrict__ kld_w,
                                               float* __restrict__ out) {
  __shared__ float sh[256];
  int tid = threadIdx.x;
  float a = 0.f;
  for (int i = tid; i < M; i += 256) a += ce[i];
  sh[tid] = a;
  __syncthreads();
  for (int o = 128; o; o >>= 1) {
    if (tid < o) sh[tid] += sh[tid + o];
    __syncthreads();
  }
  if (tid == 0) {
    float rec = sh[0] / (float)M;
    float klh = kl[0], klc = kl[1];
    out[0] = rec + (klh + klc) * kld_w[0];
    out[1] = rec;
    out[2] = klh;
    out[3] = klc;
  }
}

// ---------------- host orchestration ----------------
extern "C" void kernel_launch(void* const* d_in, const int* in_sizes, int n_in,
                              void* d_out, int out_size, void* d_ws, size_t ws_size,
                              hipStream_t stream) {
  (void)in_sizes; (void)n_in; (void)out_size; (void)ws_size;
  const int*   tokens    = (const int*)d_in[0];
  const int*   condition = (const int*)d_in[1];
  const float* eps_h     = (const float*)d_in[2];
  const float* eps_c     = (const float*)d_in[3];
  const float* kld_w     = (const float*)d_in[4];
  const float* enc_embed = (const float*)d_in[5];
  const float* enc_cond  = (const float*)d_in[6];
  const float* enc_W_ih  = (const float*)d_in[7];
  const float* enc_W_hh  = (const float*)d_in[8];
  const float* enc_b_ih  = (const float*)d_in[9];
  const float* enc_b_hh  = (const float*)d_in[10];
  const float* fc_mh_w   = (const float*)d_in[11];
  const float* fc_mh_b   = (const float*)d_in[12];
  const float* fc_lh_w   = (const float*)d_in[13];
  const float* fc_lh_b   = (const float*)d_in[14];
  const float* fc_mc_w   = (const float*)d_in[15];
  const float* fc_mc_b   = (const float*)d_in[16];
  const float* fc_lc_w   = (const float*)d_in[17];
  const float* fc_lc_b   = (const float*)d_in[18];
  const float* dec_embed = (const float*)d_in[19];
  const float* dec_cond  = (const float*)d_in[20];
  const float* dec_W_ih  = (const float*)d_in[21];
  const float* dec_W_hh  = (const float*)d_in[22];
  const float* dec_b_ih  = (const float*)d_in[23];
  const float* dec_b_hh  = (const float*)d_in[24];
  const float* fc_h_w    = (const float*)d_in[25];
  const float* fc_h_b    = (const float*)d_in[26];
  const float* fc_c_w    = (const float*)d_in[27];
  const float* fc_c_b    = (const float*)d_in[28];
  const float* out_w     = (const float*)d_in[29];
  const float* out_b     = (const float*)d_in[30];
  float* out = (float*)d_out;

  // workspace bump allocator (256B aligned)
  char* wp = (char*)d_ws;
  auto alloc = [&](size_t bytes) -> char* {
    char* p = wp;
    wp += (bytes + 255) & ~(size_t)255;
    return p;
  };
  bf16*  Xenc = (bf16*)alloc((size_t)TENC * HCD * 2);
  bf16*  Xdec = (bf16*)alloc((size_t)TDEC * HCD * 2);
  bf16*  WihE = (bf16*)alloc((size_t)G4 * HCD * 2);
  bf16*  WihD = (bf16*)alloc((size_t)G4 * HCD * 2);
  bf16*  OutW = (bf16*)alloc((size_t)VOC * HCD * 2);
  float* preE = (float*)alloc((size_t)TENC * G4 * 4);
  float* preD = (float*)alloc((size_t)TDEC * G4 * 4);
  float* h_g  = (float*)alloc(2 * HCD * 4);   // encoder ping-pong h
  float* c_g  = (float*)alloc(HCD * 4);       // encoder final c
  float* h2   = (float*)alloc(HID * 4);
  float* c2   = (float*)alloc(HID * 4);
  float* proj = (float*)alloc(4 * HID * 4);
  float* z_h  = (float*)alloc(HID * 4);
  float* z_c  = (float*)alloc(HID * 4);
  float* kl   = (float*)alloc(2 * 4);
  float* hn   = (float*)alloc(HCD * 4);
  float* cn   = (float*)alloc(HCD * 4);
  bf16*  Hall = (bf16*)alloc((size_t)TDEC * HCD * 2);
  float* chM  = (float*)alloc((size_t)TDEC * NCHUNK * 4);
  float* chS  = (float*)alloc((size_t)TDEC * NCHUNK * 4);
  float* tgtl = (float*)alloc((size_t)TDEC * 4);
  float* ce   = (float*)alloc((size_t)TDEC * 4);
  int*   bar  = (int*)alloc(2 * 4);

  // 0) reset grid-barrier state (every call; graph-capture safe)
  k_zero_i32<<<1, 32, 0, stream>>>(bar, 2);

  // 1) weight + activation conversion to bf16
  k_f32_to_bf16<<<4096, 256, 0, stream>>>(enc_W_ih, WihE, (size_t)G4 * HCD);
  k_f32_to_bf16<<<4096, 256, 0, stream>>>(dec_W_ih, WihD, (size_t)G4 * HCD);
  k_f32_to_bf16<<<8192, 256, 0, stream>>>(out_w, OutW, (size_t)VOC * HCD);
  k_gather_embed<<<4096, 256, 0, stream>>>(enc_embed, tokens, 1, Xenc, TENC);
  k_gather_embed<<<4096, 256, 0, stream>>>(dec_embed, tokens, 0, Xdec, TDEC);

  // 2) batched input projections (WMMA): pre = X @ W_ih^T + b_ih + b_hh
  //    tiles: ceil(M/16) x (N/64) tasks, one per wave
  k_wmma_gemm_bias<<<1024, 256, 0, stream>>>(Xenc, WihE, enc_b_ih, enc_b_hh, preE, TENC, G4, HCD);
  k_wmma_gemm_bias<<<1024, 256, 0, stream>>>(Xdec, WihD, dec_b_ih, dec_b_hh, preD, TDEC, G4, HCD);

  // 3) encoder recurrence (persistent kernel, device-wide barrier)
  k_lstm_encoder<<<NWGREC, 256, 0, stream>>>(enc_W_hh, preE, enc_cond, condition,
                                             h_g, c_g, TENC, bar);
  float* hT = h_g + ((TENC & 1) ? HCD : 0);   // final h buffer

  // 4) VAE head: m/lv projections, reparameterize, KL
  k_fc4<<<128, 256, 0, stream>>>(hT, c_g,
                                 fc_mh_w, fc_mh_b, fc_lh_w, fc_lh_b,
                                 fc_mc_w, fc_mc_b, fc_lc_w, fc_lc_b, proj);
  k_z_kl<<<1, 256, 0, stream>>>(proj, eps_h, eps_c, z_h, z_c, kl);

  // 5) decoder recurrence (persistent kernel), stores h_new per step as bf16
  k_lstm_decoder<<<NWGREC, 256, 0, stream>>>(dec_W_hh, preD,
                                             fc_h_w, fc_h_b, fc_c_w, fc_c_b,
                                             dec_cond, condition, z_h, z_c,
                                             h2, c2, hn, cn, Hall, TDEC, bar);

  // 6) big logits GEMM (WMMA) fused with chunked logsumexp stats
  //    128 row-tiles x 20 chunks = 2560 wave tasks
  k_wmma_logits_lse<<<320, 256, 0, stream>>>(Hall, OutW, out_b, chM, chS, TDEC, VOC, HCD);

  // 7) target logits, per-step CE, final loss
  k_tgt_logit<<<64, 256, 0, stream>>>(Hall, OutW, out_b, tokens, tgtl, TDEC);
  k_ce<<<(TDEC + 255) / 256, 256, 0, stream>>>(chM, chS, tgtl, ce, TDEC);
  k_final<<<1, 256, 0, stream>>>(ce, TDEC, kl, kld_w, out);
}
